// ReEncoder_65352222376051
// MI455X (gfx1250) — matrix operand order repaired and checked by reading, not compiled, not run on hardware
//
#include <hip/hip_runtime.h>

typedef __bf16 bf16;
typedef __attribute__((ext_vector_type(16))) __bf16 v16bf;
typedef __attribute__((ext_vector_type(8)))  __bf16 v8bf;
typedef __attribute__((ext_vector_type(8)))  float   v8f;
typedef __attribute__((ext_vector_type(4)))  unsigned int u32x4;
typedef __attribute__((ext_vector_type(8)))  int i32x8;
typedef __attribute__((ext_vector_type(4)))  int i32x4;

__device__ __constant__ int d_CH[4] = {64, 128, 256, 512};

// -------------------------------------------------------------------------
// Elementwise helpers
// -------------------------------------------------------------------------

__global__ void f32_to_bf16_kernel(const float* __restrict__ src,
                                   bf16* __restrict__ dst, long long total) {
  long long i = (long long)blockIdx.x * blockDim.x + threadIdx.x;
  if (i < total) dst[i] = (bf16)src[i];
}

// OIHW fp32 [Cout,Cin,3,3] -> bf16 [Cout, tap(9), Cin]
__global__ void reorder_w_bf16_kernel(const float* __restrict__ src,
                                      bf16* __restrict__ dst,
                                      int Cout, int Cin, long long total) {
  long long i = (long long)blockIdx.x * blockDim.x + threadIdx.x;
  if (i >= total) return;
  int c = (int)(i % Cin);
  int t = (int)((i / Cin) % 9);
  int o = (int)(i / ((long long)Cin * 9));
  dst[i] = (bf16)src[((long long)o * Cin + c) * 9 + t];
}

// f(NHWC) = bf16(prior_half * z(NCHW))
__global__ void premul_bf16_kernel(const float* __restrict__ z,
                                   const float* __restrict__ m,
                                   bf16* __restrict__ f,
                                   int C, int HW, long long total) {
  long long i = (long long)blockIdx.x * blockDim.x + threadIdx.x;
  if (i >= total) return;
  int c   = (int)(i % C);
  int pix = (int)((i / C) % HW);
  int n   = (int)(i / ((long long)C * HW));
  f[i] = (bf16)(z[((long long)n * C + c) * HW + pix] * m[(long long)n * HW + pix]);
}

// Single-channel prior resize: ac=0 -> half-pixel, ac=1 -> align_corners
__global__ void resize_prior_kernel(const float* __restrict__ src,
                                    float* __restrict__ dst,
                                    int Hs, int Ws, int Hd, int Wd,
                                    int ac, long long total) {
  long long i = (long long)blockIdx.x * blockDim.x + threadIdx.x;
  if (i >= total) return;
  int x = (int)(i % Wd);
  int y = (int)((i / Wd) % Hd);
  int n = (int)(i / ((long long)Wd * Hd));
  float sy, sx;
  if (ac) {
    sy = (Hd > 1) ? (float)y * (float)(Hs - 1) / (float)(Hd - 1) : 0.0f;
    sx = (Wd > 1) ? (float)x * (float)(Ws - 1) / (float)(Wd - 1) : 0.0f;
  } else {
    sy = ((float)y + 0.5f) * (float)Hs / (float)Hd - 0.5f;
    sx = ((float)x + 0.5f) * (float)Ws / (float)Wd - 0.5f;
    sy = fminf(fmaxf(sy, 0.0f), (float)(Hs - 1));
    sx = fminf(fmaxf(sx, 0.0f), (float)(Ws - 1));
  }
  int y0 = (int)sy, x0 = (int)sx;
  int y1 = min(y0 + 1, Hs - 1), x1 = min(x0 + 1, Ws - 1);
  float wy = sy - (float)y0, wx = sx - (float)x0;
  const float* s = src + (long long)n * Hs * Ws;
  float v00 = s[(long long)y0 * Ws + x0], v01 = s[(long long)y0 * Ws + x1];
  float v10 = s[(long long)y1 * Ws + x0], v11 = s[(long long)y1 * Ws + x1];
  dst[i] = (v00 * (1.0f - wx) + v01 * wx) * (1.0f - wy) +
           (v10 * (1.0f - wx) + v11 * wx) * wy;
}

// NHWC bf16 align-corners bilinear resize
__global__ void resize_ac_bf16_kernel(const bf16* __restrict__ src,
                                      bf16* __restrict__ dst,
                                      int C, int Hs, int Ws, int Hd, int Wd,
                                      long long total) {
  long long i = (long long)blockIdx.x * blockDim.x + threadIdx.x;
  if (i >= total) return;
  int c = (int)(i % C);
  int x = (int)((i / C) % Wd);
  int y = (int)((i / ((long long)C * Wd)) % Hd);
  int n = (int)(i / ((long long)C * Wd * Hd));
  float sy = (Hd > 1) ? (float)y * (float)(Hs - 1) / (float)(Hd - 1) : 0.0f;
  float sx = (Wd > 1) ? (float)x * (float)(Ws - 1) / (float)(Wd - 1) : 0.0f;
  int y0 = (int)sy, x0 = (int)sx;
  int y1 = min(y0 + 1, Hs - 1), x1 = min(x0 + 1, Ws - 1);
  float wy = sy - (float)y0, wx = sx - (float)x0;
  const bf16* s = src + (long long)n * Hs * Ws * C;
  float v00 = (float)s[((long long)y0 * Ws + x0) * C + c];
  float v01 = (float)s[((long long)y0 * Ws + x1) * C + c];
  float v10 = (float)s[((long long)y1 * Ws + x0) * C + c];
  float v11 = (float)s[((long long)y1 * Ws + x1) * C + c];
  float v = (v00 * (1.0f - wx) + v01 * wx) * (1.0f - wy) +
            (v10 * (1.0f - wx) + v11 * wx) * wy;
  dst[i] = (bf16)v;
}

// fused(NCHW) += att[n,kq,jj] * resize_ac(src(NCHW) -> (Hd,Wd))
__global__ void accum_resize_kernel(const float* __restrict__ src,
                                    float* __restrict__ dst,
                                    const float* __restrict__ att,
                                    int kq, int jj, int C,
                                    int Hs, int Ws, int Hd, int Wd,
                                    long long total) {
  long long i = (long long)blockIdx.x * blockDim.x + threadIdx.x;
  if (i >= total) return;
  int x = (int)(i % Wd);
  int y = (int)((i / Wd) % Hd);
  int c = (int)((i / ((long long)Wd * Hd)) % C);
  int n = (int)(i / ((long long)Wd * Hd * C));
  float w = att[n * 16 + kq * 4 + jj];
  float v;
  if (Hs == Hd && Ws == Wd) {
    v = src[i];
  } else {
    float sy = (Hd > 1) ? (float)y * (float)(Hs - 1) / (float)(Hd - 1) : 0.0f;
    float sx = (Wd > 1) ? (float)x * (float)(Ws - 1) / (float)(Wd - 1) : 0.0f;
    int y0 = (int)sy, x0 = (int)sx;
    int y1 = min(y0 + 1, Hs - 1), x1 = min(x0 + 1, Ws - 1);
    float wy = sy - (float)y0, wx = sx - (float)x0;
    const float* s = src + ((long long)n * C + c) * Hs * Ws;
    float v00 = s[(long long)y0 * Ws + x0], v01 = s[(long long)y0 * Ws + x1];
    float v10 = s[(long long)y1 * Ws + x0], v11 = s[(long long)y1 * Ws + x1];
    v = (v00 * (1.0f - wx) + v01 * wx) * (1.0f - wy) +
        (v10 * (1.0f - wx) + v11 * wx) * wy;
  }
  dst[i] += w * v;
}

// NCHW fp32 -> NHWC bf16
__global__ void nchw_to_nhwc_bf16_kernel(const float* __restrict__ src,
                                         bf16* __restrict__ dst,
                                         int C, int HW, long long total) {
  long long i = (long long)blockIdx.x * blockDim.x + threadIdx.x;
  if (i >= total) return;
  int c   = (int)(i % C);
  int pix = (int)((i / C) % HW);
  int n   = (int)(i / ((long long)C * HW));
  dst[i] = (bf16)src[((long long)n * C + c) * HW + pix];
}

// -------------------------------------------------------------------------
// Masked global average pool: pooled[n,c] = sum(z*m)/sum(m)   (z NCHW)
// -------------------------------------------------------------------------
__global__ __launch_bounds__(256)
void masked_pool_kernel(const float* __restrict__ z,
                        const float* __restrict__ m,
                        float* __restrict__ pooled, int C, int HW) {
  int c = blockIdx.x, n = blockIdx.y, tid = threadIdx.x;
  const float* zp = z + ((long long)n * C + c) * HW;
  const float* mp = m + (long long)n * HW;
  float szm = 0.0f, sm = 0.0f;
  for (int i = tid; i < HW; i += 256) {
    float mv = mp[i];
    szm += zp[i] * mv;
    sm  += mv;
  }
  __shared__ float s1[256], s2[256];
  s1[tid] = szm; s2[tid] = sm;
  __syncthreads();
  for (int o = 128; o > 0; o >>= 1) {
    if (tid < o) { s1[tid] += s1[tid + o]; s2[tid] += s2[tid + o]; }
    __syncthreads();
  }
  if (tid == 0) pooled[(long long)n * C + c] = s1[0] / s2[0];
}

// -------------------------------------------------------------------------
// q/k linears + 4x4 softmax attention (one block, N=4, BASE=32)
// -------------------------------------------------------------------------
struct QKArgs {
  const float* pooled[4];
  const float* Wq[4]; const float* bq[4];
  const float* Wk[4]; const float* bk[4];
  float* att;   // [N*16]
};

__global__ __launch_bounds__(256)
void qk_attention_kernel(QKArgs a) {
  __shared__ float qv[4][4][32];
  __shared__ float kv[4][4][32];
  __shared__ float sc[4][4][4];
  int tid = threadIdx.x;
  for (int t = tid; t < 2 * 4 * 4 * 32; t += 256) {
    int which = t / 512;
    int r = t % 512;
    int n = r / 128;
    int j = (r / 32) % 4;
    int d = r % 32;
    int Cj = d_CH[j];
    const float* W = which ? a.Wk[j] : a.Wq[j];
    const float* b = which ? a.bk[j] : a.bq[j];
    const float* p = a.pooled[j] + (long long)n * Cj;
    float acc = b[d];
    for (int c = 0; c < Cj; ++c) acc += W[d * Cj + c] * p[c];
    if (which) kv[n][j][d] = acc; else qv[n][j][d] = acc;
  }
  __syncthreads();
  if (tid < 64) {
    int n = tid / 16, qi = (tid / 4) % 4, ki = tid % 4;
    float acc = 0.0f;
    for (int d = 0; d < 32; ++d) acc += qv[n][qi][d] * kv[n][ki][d];
    sc[n][qi][ki] = acc * 0.17677669529663687f;  // 1/sqrt(32)
  }
  __syncthreads();
  if (tid < 16) {
    int n = tid / 4, qi = tid % 4;
    float mx = sc[n][qi][0];
    for (int ki = 1; ki < 4; ++ki) mx = fmaxf(mx, sc[n][qi][ki]);
    float e[4], s = 0.0f;
    for (int ki = 0; ki < 4; ++ki) { e[ki] = __expf(sc[n][qi][ki] - mx); s += e[ki]; }
    for (int ki = 0; ki < 4; ++ki) a.att[n * 16 + qi * 4 + ki] = e[ki] / s;
  }
}

// -------------------------------------------------------------------------
// Implicit-GEMM conv via v_wmma_f32_16x16x32_bf16, NHWC input, with the
// block's 32x64 weight tile staged into LDS by the Tensor Data Mover.
//   GEMM: D[Cout, HW] = Wt[Cout, K] x im2col[K, HW],  K = (tap, cin)
//   Input  : NHWC bf16 [N, H, W, Cin]
//   Weights: bf16 [Cout, KH*KH, Cin]
//   Output : NCHW fp32 [N, Cout, H, W]
//   block = 128 (4 waves); wave w owns spatial cols [bx*64+w*16, +16).
//   Wave computes 32 Cout rows x 16 cols (a0/a1 from LDS share one B frag).
//   K is always a multiple of 64 here; TDM stages 32 rows x 64 K per step,
//   padded to 144B row stride (pad_interval=32 dwords, pad_amount=4 dwords).
// mode 0: out = relu(acc*gamma/sqrt(1+eps) + beta)   (p0=gamma, p1=beta)
// mode 1: out = acc + bias + z                        (p0=bias,  p1=z NCHW)
// -------------------------------------------------------------------------
#define LDSA_STRIDE 72   // bf16 elements per staged row (64 data + 8 pad)

__global__ __launch_bounds__(128)
void conv_wmma_kernel(const bf16* __restrict__ in,
                      const bf16* __restrict__ wt,
                      const float* __restrict__ p0,
                      const float* __restrict__ p1,
                      float* __restrict__ out,
                      int Cin, int Cout, int H, int W, int KH, int mode) {
  const int HW   = H * W;
  const int K    = Cin * KH * KH;
  const int lane = threadIdx.x & 31;
  const int wave = threadIdx.x >> 5;
  const int grp  = lane >> 4;
  const int l16  = lane & 15;
  const int m0   = blockIdx.y * 32;
  const int sp   = blockIdx.x * 64 + wave * 16 + l16;
  const int batch = blockIdx.z;
  const int y = sp / W, x = sp - (sp / W) * W;

  const bf16* inB = in + (long long)batch * HW * Cin;

  __shared__ __align__(16) bf16 ldsA[32 * LDSA_STRIDE];
  // Flat shared-aperture pointers carry the LDS byte offset in addr[31:0].
  const unsigned int ldsOff = (unsigned int)(unsigned long long)&ldsA[0];

  // Static part of the TDM descriptor (group 1):
  //   data_size=1 (2B), pad_enable=1, pad_interval=4 (32 dwords),
  //   pad_amount=3 (4 dwords), tensor 64 x 32, tile 64 x 32, dim0 stride = K.
  i32x8 g1;
  g1[0] = 0x07110000;
  g1[1] = (64 << 16);   // tensor_dim0[15:0]
  g1[2] = (32 << 16);   // tensor_dim0[31:16]=0 | tensor_dim1[15:0]
  g1[3] = (64 << 16);   // tensor_dim1[31:16]=0 | tile_dim0
  g1[4] = 32;           // tile_dim1 (tile_dim2 = 0)
  g1[5] = K;            // tensor_dim0_stride[31:0]
  g1[6] = 0;
  g1[7] = 0;
  const i32x4 gz4 = {0, 0, 0, 0};
  const i32x8 gz8 = {0, 0, 0, 0, 0, 0, 0, 0};

  v8f acc0 = {}, acc1 = {};
  for (int k0 = 0; k0 < K; k0 += 64) {
    if (wave == 0) {
      unsigned long long ga =
          (unsigned long long)(wt + (long long)m0 * K + k0);
      u32x4 g0;
      g0[0] = 1u;                      // count=1, user descriptor
      g0[1] = ldsOff;                  // lds_addr
      g0[2] = (unsigned int)ga;        // global_addr[31:0]
      g0[3] = (unsigned int)((ga >> 32) & 0x1FFFFFFull) | 0x80000000u;  // type=2
      __builtin_amdgcn_tensor_load_to_lds(g0, g1, gz4, gz4, gz8, 0);
      __builtin_amdgcn_s_wait_tensorcnt(0);
    }
    __syncthreads();

#pragma unroll
    for (int c0 = 0; c0 < 64; c0 += 32) {
      // ---- A fragments from LDS. k order used here (same as B): lanes 0-15 K[0..15],
      //      lanes 16-31 K[16..31] -> two 16B ds loads per lane.
      v16bf a0, a1;
      {
        const bf16* r0 = &ldsA[l16 * LDSA_STRIDE + c0];
        const bf16* r1 = &ldsA[(l16 + 16) * LDSA_STRIDE + c0];
        v8bf lo0 = *(const v8bf*)(r0 + 16 * grp);
        v8bf hi0 = *(const v8bf*)(r0 + 16 * grp + 8);
        v8bf lo1 = *(const v8bf*)(r1 + 16 * grp);
        v8bf hi1 = *(const v8bf*)(r1 + 16 * grp + 8);
#pragma unroll
        for (int e = 0; e < 8; ++e) {
          a0[e] = lo0[e]; a0[8 + e] = hi0[e];
          a1[e] = lo1[e]; a1[8 + e] = hi1[e];
        }
      }
      // ---- B fragment: lane = column, K run = [k0+c0+16*grp, +16) lies
      //      inside a single 3x3 tap (16 | Cin) -> 16 contiguous bf16 (NHWC).
      const int kk0 = k0 + c0 + 16 * grp;
      const bf16* bp;
      bool valid;
      if (KH == 3) {
        int tap  = kk0 / Cin;
        int coff = kk0 - tap * Cin;
        int r = tap / 3, s_ = tap - r * 3;
        int yy = y + r - 1, xx = x + s_ - 1;
        valid = (yy >= 0 && yy < H && xx >= 0 && xx < W);
        bp = inB + ((long long)yy * W + xx) * Cin + coff;
      } else {
        valid = true;
        bp = inB + (long long)sp * Cin + kk0;
      }
      v16bf b;
      if (valid) {
        v8bf blo = *(const v8bf*)bp;
        v8bf bhi = *(const v8bf*)(bp + 8);
#pragma unroll
        for (int e = 0; e < 8; ++e) { b[e] = blo[e]; b[8 + e] = bhi[e]; }
      } else {
#pragma unroll
        for (int e = 0; e < 16; ++e) b[e] = (bf16)0.0f;
      }
      acc0 = __builtin_amdgcn_wmma_f32_16x16x32_bf16(
          false, a0, false, b, (short)0, acc0, false, false);
      acc1 = __builtin_amdgcn_wmma_f32_16x16x32_bf16(
          false, a1, false, b, (short)0, acc1, false, false);
    }
    __syncthreads();   // protect ldsA before next TDM overwrite
  }

  // ---- epilogue: C/D layout -> lane col = l16, VGPR v -> row v + 8*grp
#pragma unroll
  for (int v = 0; v < 8; ++v) {
#pragma unroll
    for (int half = 0; half < 2; ++half) {
      int m = m0 + half * 16 + v + 8 * grp;
      long long oidx = ((long long)batch * Cout + m) * HW + sp;
      float val = half ? acc1[v] : acc0[v];
      if (mode == 0) {
        val = val * (p0[m] * 0.9999950000374997f) + p1[m];  // gamma/sqrt(1+1e-5)
        out[oidx] = fmaxf(val, 0.0f);
      } else {
        out[oidx] = val + p0[m] + p1[oidx];
      }
    }
  }
}

// -------------------------------------------------------------------------
// Host orchestration
// -------------------------------------------------------------------------
static inline size_t alignUp256(size_t v) { return (v + 255) & ~(size_t)255; }
static inline int nblk(long long total) { return (int)((total + 255) / 256); }

extern "C" void kernel_launch(void* const* d_in, const int* in_sizes, int n_in,
                              void* d_out, int out_size, void* d_ws, size_t ws_size,
                              hipStream_t stream) {
  (void)in_sizes; (void)n_in; (void)out_size; (void)ws_size;
  const int CH_[4] = {64, 128, 256, 512};
  const int HS_[4] = {128, 64, 32, 16};
  const int HW_[4] = {16384, 4096, 1024, 256};
  const int Nb = 4;

  const float* z[4] = {(const float*)d_in[0], (const float*)d_in[1],
                       (const float*)d_in[2], (const float*)d_in[3]};
  const float* prior[2] = {(const float*)d_in[4], (const float*)d_in[5]};  // fg, bg
  const float* P = (const float*)d_in[7];  // params blob (pytree leaves, sorted keys)

  // ---------------- param offsets (floats), alphabetical key order ----------
  // ff_bg, ff_fg, k_bg, k_fg, q_bg, q_fg, sc_bg, sc_fg ; side 0=fg 1=bg
  size_t ffW[2][4][4], ffG[2][4][4], ffB[2][4][4];
  size_t kW[2][4], kB[2][4], qW[2][4], qB[2][4], scW[2][4], scB[2][4];
  size_t off = 0;
  for (int s_ = 0; s_ < 2; ++s_) {
    int side = (s_ == 0) ? 1 : 0;
    for (int k = 0; k < 4; ++k)
      for (int j = 0; j < 4; ++j) {
        ffW[side][k][j] = off; off += (size_t)CH_[k] * CH_[j] * 9;
        ffG[side][k][j] = off; off += CH_[k];
        ffB[side][k][j] = off; off += CH_[k];
      }
  }
  for (int s_ = 0; s_ < 2; ++s_) {
    int side = (s_ == 0) ? 1 : 0;
    for (int j = 0; j < 4; ++j) { kW[side][j] = off; off += 32 * (size_t)CH_[j];
                                  kB[side][j] = off; off += 32; }
  }
  for (int s_ = 0; s_ < 2; ++s_) {
    int side = (s_ == 0) ? 1 : 0;
    for (int j = 0; j < 4; ++j) { qW[side][j] = off; off += 32 * (size_t)CH_[j];
                                  qB[side][j] = off; off += 32; }
  }
  for (int s_ = 0; s_ < 2; ++s_) {
    int side = (s_ == 0) ? 1 : 0;
    for (int k = 0; k < 4; ++k) { scW[side][k] = off; off += (size_t)CH_[k] * CH_[k];
                                  scB[side][k] = off; off += CH_[k]; }
  }

  // ---------------- workspace layout ----------------
  char* ws = (char*)d_ws;
  size_t o = 0;
  auto alloc = [&](size_t bytes) -> char* {
    char* p = ws + o; o = alignUp256(o + bytes); return p;
  };
  float* p_half[2][4]; float* p_ac[2][4]; float* pooled[2][4]; float* att[2];
  bf16* fbuf[2][4]; bf16* fdown[2][4][4]; float* fused[2][4]; bf16* fusedb[2][4];
  bf16* wff[2][4][4]; bf16* wsc[2][4];
  for (int s = 0; s < 2; ++s) {
    for (int j = 0; j < 4; ++j) {
      p_half[s][j] = (float*)alloc((size_t)Nb * HW_[j] * 4);
      p_ac[s][j]   = (float*)alloc((size_t)Nb * HW_[j] * 4);
      pooled[s][j] = (float*)alloc((size_t)Nb * CH_[j] * 4);
    }
    att[s] = (float*)alloc((size_t)Nb * 16 * 4);
    for (int j = 0; j < 4; ++j)
      fbuf[s][j] = (bf16*)alloc((size_t)Nb * CH_[j] * HW_[j] * 2);
    for (int k = 0; k < 4; ++k)
      for (int j = 0; j < k; ++j)
        fdown[s][k][j] = (bf16*)alloc((size_t)Nb * CH_[j] * HW_[k] * 2);
    for (int k = 0; k < 4; ++k) {
      fused[s][k]  = (float*)alloc((size_t)Nb * CH_[k] * HW_[k] * 4);
      fusedb[s][k] = (bf16*)alloc((size_t)Nb * CH_[k] * HW_[k] * 2);
    }
    for (int k = 0; k < 4; ++k)
      for (int j = 0; j < 4; ++j)
        wff[s][k][j] = (bf16*)alloc((size_t)CH_[k] * CH_[j] * 9 * 2);
    for (int k = 0; k < 4; ++k)
      wsc[s][k] = (bf16*)alloc((size_t)CH_[k] * CH_[k] * 2);
  }
  float* convtmp = (float*)alloc((size_t)Nb * 64 * 16384 * 4);  // max conv out

  // ---------------- output layout ----------------
  float* outp = (float*)d_out;
  float* out_fg[4]; float* out_bg[4];
  size_t oo = 0;
  for (int k = 0; k < 4; ++k) { out_fg[k] = outp + oo; oo += (size_t)Nb * CH_[k] * HW_[k]; }
  for (int k = 0; k < 4; ++k) { out_bg[k] = outp + oo; oo += (size_t)Nb * CH_[k] * HW_[k]; }
  float* out_p1 = outp + oo; oo += (size_t)Nb * 512 * 512;
  float* out_p2 = outp + oo;

  // ---------------- weight conversion / reorder fp32 -> bf16 ----------------
  for (int s = 0; s < 2; ++s) {
    for (int k = 0; k < 4; ++k)
      for (int j = 0; j < 4; ++j) {
        long long cnt = (long long)CH_[k] * CH_[j] * 9;
        reorder_w_bf16_kernel<<<nblk(cnt), 256, 0, stream>>>(
            P + ffW[s][k][j], wff[s][k][j], CH_[k], CH_[j], cnt);
      }
    for (int k = 0; k < 4; ++k) {
      long long cnt = (long long)CH_[k] * CH_[k];
      f32_to_bf16_kernel<<<nblk(cnt), 256, 0, stream>>>(P + scW[s][k], wsc[s][k], cnt);
    }
  }

  // ---------------- main pipeline ----------------
  for (int s = 0; s < 2; ++s) {
    // prior resizes
    for (int j = 0; j < 4; ++j) {
      long long tot = (long long)Nb * HW_[j];
      resize_prior_kernel<<<nblk(tot), 256, 0, stream>>>(
          prior[s], p_half[s][j], 512, 512, HS_[j], HS_[j], 0, tot);
      resize_prior_kernel<<<nblk(tot), 256, 0, stream>>>(
          prior[s], p_ac[s][j], 512, 512, HS_[j], HS_[j], 1, tot);
    }
    // f(NHWC bf16) = p_half * z
    for (int j = 0; j < 4; ++j) {
      long long tot = (long long)Nb * CH_[j] * HW_[j];
      premul_bf16_kernel<<<nblk(tot), 256, 0, stream>>>(
          z[j], p_half[s][j], fbuf[s][j], CH_[j], HW_[j], tot);
    }
    // masked pooling
    for (int j = 0; j < 4; ++j)
      masked_pool_kernel<<<dim3(CH_[j], Nb), 256, 0, stream>>>(
          z[j], p_ac[s][j], pooled[s][j], CH_[j], HW_[j]);
    // attention
    QKArgs qa;
    for (int j = 0; j < 4; ++j) {
      qa.pooled[j] = pooled[s][j];
      qa.Wq[j] = P + qW[s][j]; qa.bq[j] = P + qB[s][j];
      qa.Wk[j] = P + kW[s][j]; qa.bk[j] = P + kB[s][j];
    }
    qa.att = att[s];
    qk_attention_kernel<<<1, 256, 0, stream>>>(qa);
    // downsampled f (NHWC) for j < k
    for (int k = 0; k < 4; ++k)
      for (int j = 0; j < k; ++j) {
        long long tot = (long long)Nb * CH_[j] * HW_[k];
        resize_ac_bf16_kernel<<<nblk(tot), 256, 0, stream>>>(
            fbuf[s][j], fdown[s][k][j], CH_[j], HS_[j], HS_[j], HS_[k], HS_[k], tot);
      }
    // fusion
    for (int k = 0; k < 4; ++k)
      (void)hipMemsetAsync(fused[s][k], 0, (size_t)Nb * CH_[k] * HW_[k] * 4, stream);
    for (int k = 0; k < 4; ++k) {
      for (int j = 0; j < 4; ++j) {
        const bf16* cin = (j < k) ? fdown[s][k][j] : fbuf[s][j];
        int Hc = (j > k) ? HS_[j] : HS_[k];
        dim3 grid(Hc * Hc / 64, CH_[k] / 32, Nb);
        conv_wmma_kernel<<<grid, 128, 0, stream>>>(
            cin, wff[s][k][j], P + ffG[s][k][j], P + ffB[s][k][j], convtmp,
            CH_[j], CH_[k], Hc, Hc, 3, 0);
        long long tot = (long long)Nb * CH_[k] * HW_[k];
        accum_resize_kernel<<<nblk(tot), 256, 0, stream>>>(
            convtmp, fused[s][k], att[s], k, j, CH_[k], Hc, Hc, HS_[k], HS_[k], tot);
      }
    }
    // score conv: out = z + conv1x1(fused) + bias
    for (int k = 0; k < 4; ++k) {
      long long tot = (long long)Nb * CH_[k] * HW_[k];
      nchw_to_nhwc_bf16_kernel<<<nblk(tot), 256, 0, stream>>>(
          fused[s][k], fusedb[s][k], CH_[k], HW_[k], tot);
      dim3 grid(HW_[k] / 64, CH_[k] / 32, Nb);
      conv_wmma_kernel<<<grid, 128, 0, stream>>>(
          fusedb[s][k], wsc[s][k], P + scB[s][k], z[k],
          (s ? out_bg : out_fg)[k], CH_[k], CH_[k], HS_[k], HS_[k], 1, 1);
    }
  }

  // trailing prior outputs (prior_fg twice)
  (void)hipMemcpyAsync(out_p1, prior[0], (size_t)Nb * 512 * 512 * 4,
                       hipMemcpyDeviceToDevice, stream);
  (void)hipMemcpyAsync(out_p2, prior[0], (size_t)Nb * 512 * 512 * 4,
                       hipMemcpyDeviceToDevice, stream);
}
